// MultichannelAttention_59768764891261
// MI455X (gfx1250) — matrix operand order repaired and checked by reading, not compiled, not run on hardware
//
#include <hip/hip_runtime.h>
#include <hip/hip_bf16.h>

typedef __attribute__((ext_vector_type(8)))  _Float16 v8h;
typedef __attribute__((ext_vector_type(16))) _Float16 v16h;
typedef __attribute__((ext_vector_type(8)))  float    v8f;
typedef __attribute__((ext_vector_type(4)))  unsigned int v4u;
typedef __attribute__((ext_vector_type(8)))  int      v8i;
typedef __attribute__((ext_vector_type(4)))  int      v4i;

#define B_   2
#define C_   8
#define H_   1024
#define W_   1024
#define ROT  128

#define BM 128
#define BN 128
#define BK 32
#define LSTR 40   // padded LDS row stride in f16 elems (80B, 16B aligned)

__device__ __forceinline__ v16h cat8(v8h lo, v8h hi) {
    return __builtin_shufflevector(lo, hi, 0,1,2,3,4,5,6,7,8,9,10,11,12,13,14,15);
}

// ---------------------------------------------------------------------------
// Batched GEMM: C[bz] = scale * A[bz?] (MxK row-major f16) @ B[bz] (KxN row-major f16)
// M = N = K = 1024. Block computes a 128x128 tile; 8 waves, each wave 64x32
// via a 4x2 grid of v_wmma_f32_16x16x32_f16 accumulators.
// A tile is staged global->LDS by the Tensor Data Mover (one descriptor per
// K-step, issued by wave 0, published with s_wait_tensorcnt + barrier); the
// TDM pad fields reproduce the 80B padded LDS row stride. B tile is staged
// with vector loads + transposed ds stores.
// ---------------------------------------------------------------------------
template<bool OUT_F16>
__global__ __launch_bounds__(256)
void gemm_bc(const _Float16* __restrict__ Abase,
             const _Float16* __restrict__ Bbase,
             void* __restrict__ Cbase,
             long aStride, int aMod, long bStride, long cStride,
             float scale)
{
    __shared__ _Float16 As[BM * LSTR];
    __shared__ _Float16 Bs[BN * LSTR];
    const int K = H_, N = W_;

    const int bz = blockIdx.z;
    const _Float16* A  = Abase + (long)(aMod ? (bz & (aMod - 1)) : bz) * aStride;
    const _Float16* Bp = Bbase + (long)bz * bStride;
    const int m0 = blockIdx.y * BM;
    const int n0 = blockIdx.x * BN;

    const int tid  = threadIdx.x;
    const int lane = tid & 31;
    const int wave = tid >> 5;
    const int wm = (wave >> 2) * 64;   // 2 wave-rows of 64
    const int wn = (wave & 3) * 32;    // 4 wave-cols of 32

    // keep As observable as "written" (TDM writes it via descriptor only)
    if ((int)blockIdx.x == -1) As[tid] = (_Float16)0;

    // B staging map: 32 k-rows x 128 n-cols, transposed into LDS (Bs[n][k])
    const int bk = tid >> 3, bn = (tid & 7) * 16;

    // ---- invariant TDM descriptor pieces (D# group 1: geometry) ----
    const unsigned ldsA = (unsigned)(unsigned long long)(void*)As;  // LDS byte offset
    v8i g1;
    g1[0] = (1 << 16)      // data_size = 1 -> 2 bytes
          | (1 << 20)      // pad_enable
          | (3 << 22)      // pad_interval: 16 DWORDs (one 64B tile row)
          | (3 << 25);     // pad_amount:   4 DWORDs (16B) -> 80B LDS stride
    g1[1] = (int)(32u  << 16);  // tensor_dim0[15:0] = 32
    g1[2] = (int)(128u << 16);  // tensor_dim0 hi = 0, tensor_dim1[15:0] = 128
    g1[3] = (int)(32u  << 16);  // tensor_dim1 hi = 0, tile_dim0 = 32
    g1[4] = 128;                // tile_dim1 = 128, tile_dim2 = 0
    g1[5] = 1024;               // tensor_dim0_stride lo = K
    g1[6] = 0;                  // stride hi / dim1_stride lo
    g1[7] = 0;
    const v4i gz4 = {0, 0, 0, 0};
    const v8i gz8 = {0, 0, 0, 0, 0, 0, 0, 0};

    v8f acc[4][2] = {};

    for (int k0 = 0; k0 < K; k0 += BK) {
        // ---- A tile via Tensor Data Mover (wave 0 only; EXEC-independent) ----
        if (wave == 0) {
            const unsigned long long ga64 =
                (unsigned long long)(const void*)(A + (long)m0 * K + k0);
            v4u g0;
            g0[0] = 1u;                                  // count = 1, user mode
            g0[1] = ldsA;                                // lds_addr
            g0[2] = (unsigned)ga64;                      // global_addr[31:0]
            g0[3] = (unsigned)((ga64 >> 32) & 0x1FFFFFFu) | (2u << 30); // [56:32] | type=2
            __builtin_amdgcn_tensor_load_to_lds(g0, g1, gz4, gz4, gz8, 0);
        }

        // ---- stage B tile (transpose on LDS store) ----
        const _Float16* gb = Bp + (long)(k0 + bk) * N + n0 + bn;
        v8h b0 = *(const v8h*)(gb);
        v8h b1 = *(const v8h*)(gb + 8);
        #pragma unroll
        for (int j = 0; j < 8; j++) Bs[(bn + j) * LSTR + bk]     = b0[j];
        #pragma unroll
        for (int j = 0; j < 8; j++) Bs[(bn + 8 + j) * LSTR + bk] = b1[j];

        if (k0 + BK < K) {
            __builtin_prefetch(gb + (long)BK * N, 0, 3);
        }

        if (wave == 0) __builtin_amdgcn_s_wait_tensorcnt(0);
        asm volatile("" ::: "memory");
        __syncthreads();

        {
            const int hh = lane >> 4;   // half-wave select
            const int ml = lane & 15;
            v16h bf[2], af[4];
            // B frag: lane=(N%16)+16*h holds K in [8h,8h+8) u [16+8h,16+8h+8) of column N
            #pragma unroll
            for (int j = 0; j < 2; j++) {
                const _Float16* p = &Bs[(wn + j * 16 + ml) * LSTR + hh * 8];
                bf[j] = cat8(*(const v8h*)p, *(const v8h*)(p + 16));
            }
            // A frag: lane=(M)+16*h holds K in [8h,8h+8) u [16+8h,16+8h+8)
            #pragma unroll
            for (int i = 0; i < 4; i++) {
                const _Float16* p = &As[(wm + i * 16 + ml) * LSTR + hh * 8];
                af[i] = cat8(*(const v8h*)p, *(const v8h*)(p + 16));
            }
            #pragma unroll
            for (int i = 0; i < 4; i++)
                #pragma unroll
                for (int j = 0; j < 2; j++)
                    acc[i][j] = __builtin_amdgcn_wmma_f32_16x16x32_f16(
                        false, af[i], false, bf[j], (short)0, acc[i][j], false, false);
        }
        __syncthreads();
    }

    // ---- epilogue: C layout — VGPR r, lane l<16: (M=r, N=l); l>=16: (M=8+r, N=l-16)
    {
        const int eh = lane >> 4;
        const int el = lane & 15;
        #pragma unroll
        for (int i = 0; i < 4; i++) {
            const int mb = m0 + wm + i * 16 + eh * 8;
            #pragma unroll
            for (int j = 0; j < 2; j++) {
                const int n = n0 + wn + j * 16 + el;
                #pragma unroll
                for (int r = 0; r < 8; r++) {
                    const float v = acc[i][j][r] * scale;
                    const long  o = (long)bz * cStride + (long)(mb + r) * N + n;
                    if (OUT_F16) ((_Float16*)Cbase)[o] = (_Float16)v;
                    else         ((float*)Cbase)[o]    = v;
                }
            }
        }
    }
}

// ---------------------------------------------------------------------------
// 3x3 SAME conv, 8->8 channels, three weight sets in one pass. f16 output.
// ---------------------------------------------------------------------------
__global__ __launch_bounds__(256)
void conv3x3_qkv(const float* __restrict__ x,
                 const float* __restrict__ wq, const float* __restrict__ wk,
                 const float* __restrict__ wv,
                 _Float16* __restrict__ yq, _Float16* __restrict__ yk,
                 _Float16* __restrict__ yv)
{
    __shared__ float sw[3][C_ * C_ * 9];
    const int tid = threadIdx.x;
    for (int i = tid; i < C_ * C_ * 9; i += 256) {
        sw[0][i] = wq[i]; sw[1][i] = wk[i]; sw[2][i] = wv[i];
    }
    __syncthreads();

    const long idx = (long)blockIdx.x * 256 + tid;
    const int  w = (int)(idx % W_);
    const long t = idx / W_;
    const int  h = (int)(t % H_);
    const int  b = (int)(t / H_);

    float aq[C_] = {}, ak[C_] = {}, av[C_] = {};
    for (int ci = 0; ci < C_; ci++) {
        const float* xp = x + (long)(b * C_ + ci) * H_ * W_;
        #pragma unroll
        for (int dh = -1; dh <= 1; dh++) {
            const int hh = h + dh;
            if (hh < 0 || hh >= H_) continue;
            #pragma unroll
            for (int dw = -1; dw <= 1; dw++) {
                const int ww = w + dw;
                if (ww < 0 || ww >= W_) continue;
                const float xv = xp[(long)hh * W_ + ww];
                const int wi = ci * 9 + (dh + 1) * 3 + (dw + 1);
                #pragma unroll
                for (int co = 0; co < C_; co++) {
                    aq[co] = fmaf(xv, sw[0][co * C_ * 9 + wi], aq[co]);
                    ak[co] = fmaf(xv, sw[1][co * C_ * 9 + wi], ak[co]);
                    av[co] = fmaf(xv, sw[2][co * C_ * 9 + wi], av[co]);
                }
            }
        }
    }
    const long base = (long)b * C_ * H_ * W_ + (long)h * W_ + w;
    #pragma unroll
    for (int co = 0; co < C_; co++) {
        yq[base + (long)co * H_ * W_] = (_Float16)aq[co];
        yk[base + (long)co * H_ * W_] = (_Float16)ak[co];
        yv[base + (long)co * H_ * W_] = (_Float16)av[co];
    }
}

// ---------------------------------------------------------------------------
// RoPE + optional transpose. src is (f, w) f16 per batch; doXpose writes (w, f).
// RoPE rotates feature pairs f<128 with position = w.
// ---------------------------------------------------------------------------
__global__ __launch_bounds__(256)
void rope_xpose(const _Float16* __restrict__ src, _Float16* __restrict__ dst,
                int doRope, int doXpose)
{
    __shared__ _Float16 tile[32][33];
    const int  bz  = blockIdx.z;
    const long off = (long)bz * (long)H_ * W_;
    const int  f0 = blockIdx.y * 32, w0 = blockIdx.x * 32;
    const int  tid = threadIdx.x;

    #pragma unroll
    for (int r = 0; r < 4; r++) {
        const int e = tid + 256 * r;
        const int ww = e & 31, ff = e >> 5;
        tile[ff][ww] = src[off + (long)(f0 + ff) * W_ + w0 + ww];
    }
    __syncthreads();

    #pragma unroll
    for (int r = 0; r < 4; r++) {
        const int e = tid + 256 * r;
        int ff, ww;
        if (doXpose) { ff = e & 31; ww = e >> 5; }
        else         { ww = e & 31; ff = e >> 5; }
        const int f = f0 + ff, pos = w0 + ww;
        float v = (float)tile[ff][ww];
        if (doRope && f < ROT) {
            const float pv   = (float)tile[ff ^ 1][ww];
            const float freq = __expf(-9.2103403719761836f * (float)(f & ~1) / (float)ROT);
            float s, c;
            __sincosf((float)pos * freq, &s, &c);
            v = (f & 1) ? (v * c + pv * s) : (v * c - pv * s);
        }
        if (doXpose) dst[off + (long)(w0 + ww) * H_ + f0 + ff] = (_Float16)v;
        else         dst[off + (long)f * W_ + w0 + ww]         = (_Float16)v;
    }
}

// ---------------------------------------------------------------------------
// Row softmax over 1024-wide f32 rows; one wave (32 lanes) per row, f16 out.
// ---------------------------------------------------------------------------
__global__ __launch_bounds__(256)
void softmax_rows(const float* __restrict__ S, _Float16* __restrict__ P)
{
    const int  lane = threadIdx.x & 31;
    const int  wave = threadIdx.x >> 5;
    const long row  = (long)blockIdx.x * 8 + wave;
    const float* sp = S + row * 1024;

    float4 v[8];
    float m = -3.0e38f;
    #pragma unroll
    for (int j = 0; j < 8; j++) {
        v[j] = *(const float4*)(sp + j * 128 + lane * 4);
        m = fmaxf(m, fmaxf(fmaxf(v[j].x, v[j].y), fmaxf(v[j].z, v[j].w)));
    }
    #pragma unroll
    for (int d = 16; d >= 1; d >>= 1) m = fmaxf(m, __shfl_xor(m, d, 32));

    float sum = 0.f;
    #pragma unroll
    for (int j = 0; j < 8; j++) {
        v[j].x = __expf(v[j].x - m); v[j].y = __expf(v[j].y - m);
        v[j].z = __expf(v[j].z - m); v[j].w = __expf(v[j].w - m);
        sum += v[j].x + v[j].y + v[j].z + v[j].w;
    }
    #pragma unroll
    for (int d = 16; d >= 1; d >>= 1) sum += __shfl_xor(sum, d, 32);
    const float inv = 1.0f / sum;

    _Float16* pp = P + row * 1024;
    #pragma unroll
    for (int j = 0; j < 8; j++) {
        const int o = j * 128 + lane * 4;
        pp[o + 0] = (_Float16)(v[j].x * inv);
        pp[o + 1] = (_Float16)(v[j].y * inv);
        pp[o + 2] = (_Float16)(v[j].z * inv);
        pp[o + 3] = (_Float16)(v[j].w * inv);
    }
}

__global__ __launch_bounds__(256)
void f32_to_f16(const float* __restrict__ s, _Float16* __restrict__ d, long n)
{
    const long i = (long)blockIdx.x * 256 + threadIdx.x;
    if (i < n) d[i] = (_Float16)s[i];
}

// ---------------------------------------------------------------------------
extern "C" void kernel_launch(void* const* d_in, const int* in_sizes, int n_in,
                              void* d_out, int out_size, void* d_ws, size_t ws_size,
                              hipStream_t stream)
{
    const float* x   = (const float*)d_in[0];
    const float* cq  = (const float*)d_in[1];
    const float* ck  = (const float*)d_in[2];
    const float* cv  = (const float*)d_in[3];
    const float* pwq = (const float*)d_in[4];
    const float* pwk = (const float*)d_in[5];
    const float* pwv = (const float*)d_in[6];
    const float* pwo = (const float*)d_in[7];

    char* ws = (char*)d_ws;
    const size_t MB  = 1ull << 20;
    const long   HHl = (long)H_ * W_;   // 1M elems per (b,c) matrix

    _Float16* pw16q = (_Float16*)(ws + 0 * MB);
    _Float16* pw16k = (_Float16*)(ws + 16 * MB);
    _Float16* pw16v = (_Float16*)(ws + 32 * MB);
    _Float16* pw16o = (_Float16*)(ws + 48 * MB);
    _Float16* Yq = (_Float16*)(ws + 64 * MB);
    _Float16* Yk = (_Float16*)(ws + 96 * MB);
    _Float16* Yv = (_Float16*)(ws + 128 * MB);
    _Float16* Pq = (_Float16*)(ws + 160 * MB);
    _Float16* Pk = (_Float16*)(ws + 192 * MB);
    _Float16* Pv = (_Float16*)(ws + 224 * MB);
    _Float16* Q  = Yq;                         // reuse after Y consumed
    _Float16* Kt = Yk;
    _Float16* V  = Yv;
    float*    S  = (float*)(ws + 160 * MB);    // overwrites consumed Pq+Pk
    _Float16* Pa = (_Float16*)(ws + 224 * MB); // overwrites consumed Pv
    _Float16* AV = (_Float16*)(ws + 64 * MB);  // overwrites consumed Q

    dim3 blk(256);

    // 1) pw weights -> f16
    const long npw = (long)C_ * H_ * H_;       // 8M
    dim3 gcvt((unsigned)((npw + 255) / 256));
    f32_to_f16<<<gcvt, blk, 0, stream>>>(pwq, pw16q, npw);
    f32_to_f16<<<gcvt, blk, 0, stream>>>(pwk, pw16k, npw);
    f32_to_f16<<<gcvt, blk, 0, stream>>>(pwv, pw16v, npw);
    f32_to_f16<<<gcvt, blk, 0, stream>>>(pwo, pw16o, npw);

    // 2) conv3x3 -> Yq/Yk/Yv (f16)
    conv3x3_qkv<<<dim3((unsigned)((long)B_ * H_ * W_ / 256)), blk, 0, stream>>>(
        x, cq, ck, cv, Yq, Yk, Yv);

    // 3) projections: P = pw[c] @ Y (f16 out)
    dim3 gg(H_ / BN, H_ / BM, B_ * C_);
    gemm_bc<true><<<gg, blk, 0, stream>>>(pw16q, Yq, Pq, HHl, C_, HHl, HHl, 1.0f);
    gemm_bc<true><<<gg, blk, 0, stream>>>(pw16k, Yk, Pk, HHl, C_, HHl, HHl, 1.0f);
    gemm_bc<true><<<gg, blk, 0, stream>>>(pw16v, Yv, Pv, HHl, C_, HHl, HHl, 1.0f);

    // 4) RoPE / transpose
    dim3 gr(32, 32, B_ * C_);
    rope_xpose<<<gr, blk, 0, stream>>>(Pq, Q,  1, 1);  // Q  (w,f), rope
    rope_xpose<<<gr, blk, 0, stream>>>(Pk, Kt, 1, 0);  // Kt (f,w), rope
    rope_xpose<<<gr, blk, 0, stream>>>(Pv, V,  0, 1);  // V  (w,f)

    // 5) S = Q @ Kt / 32  (f32)
    gemm_bc<false><<<gg, blk, 0, stream>>>(Q, Kt, S, HHl, 0, HHl, HHl, 0.03125f);

    // 6) softmax rows -> Pa (f16)
    softmax_rows<<<dim3((unsigned)(B_ * C_ * H_ / 8)), blk, 0, stream>>>(S, Pa);

    // 7) AV = Pa @ V (f16)
    gemm_bc<true><<<gg, blk, 0, stream>>>(Pa, V, AV, HHl, 0, HHl, HHl, 1.0f);

    // 8) out = pw_out[c] @ AV (f32 -> d_out)
    gemm_bc<false><<<gg, blk, 0, stream>>>(pw16o, AV, (float*)d_out, HHl, C_, HHl, HHl, 1.0f);
}
